// PairFormerIteration_57363583205597
// MI455X (gfx1250) — matrix operand order repaired and checked by reading, not compiled, not run on hardware
//
#include <hip/hip_runtime.h>
#include <cstdint>

typedef __bf16 bhalf;
typedef __attribute__((ext_vector_type(16))) __bf16 v16bf;
typedef __attribute__((ext_vector_type(8)))  float  v8f;

#define NSEQ 384
#define CCH  128
#define M2   147456   // NSEQ*NSEQ
#define HD   128      // H*D

__device__ inline float sigmf(float x){ return 1.0f/(1.0f + __expf(-x)); }
__device__ inline v8f vz8(){ v8f z;
#pragma unroll
  for (int i=0;i<8;i++) z[i]=0.0f; return z; }

__device__ inline v8f wmma_bf16(v16bf a, v16bf b, v8f c){
  return __builtin_amdgcn_wmma_f32_16x16x32_bf16(false, a, false, b, (short)0, c, false, false);
}

// A-operand (16x32 bf16): lane m=lane&15; K = {k0..k0+7} in elems 0-7 and {k0+16..k0+23}
// in elems 8-15, where k0 = (lane>>4)*8.  (ISA 7.12.2 16-bit A-matrix table)
__device__ inline v16bf ld_frag_a(const bhalf* base, int stride){
  int lane = threadIdx.x & 31;
  const bhalf* p = base + (size_t)(lane & 15)*stride + ((lane >> 4) << 3);
  union { uint4 u[2]; v16bf v; } x;
  x.u[0] = *reinterpret_cast<const uint4*>(p);
  x.u[1] = *reinterpret_cast<const uint4*>(p + 16);
  return x.v;
}
// B-operand (32x16 bf16): lane n=lane&15; K = {k0..k0+7} in elems 0-7 and {k0+16..k0+23}
// in elems 8-15, where k0 = (lane>>4)*8 (same K order as A).
__device__ inline v16bf ld_frag_b(const bhalf* base, int stride){
  int lane = threadIdx.x & 31;
  const bhalf* p = base + (size_t)(lane & 15)*stride + ((lane >> 4) << 3);
  union { uint4 u[2]; v16bf v; } x;
  x.u[0] = *reinterpret_cast<const uint4*>(p);
  x.u[1] = *reinterpret_cast<const uint4*>(p + 16);
  return x.v;
}

// ---------------- LayerNorm over C (last axis), f32 -> bf16 ----------------
__global__ __launch_bounds__(256) void ln_kernel(const float* __restrict__ act,
    const float* __restrict__ sc, const float* __restrict__ bi, bhalf* __restrict__ out)
{
  int t = threadIdx.x;
  int row  = blockIdx.x * 8 + (t >> 5);
  int lane = t & 31;
  const float4 xv = *reinterpret_cast<const float4*>(act + (size_t)row*CCH + lane*4);
  float s = xv.x + xv.y + xv.z + xv.w;
  float q = xv.x*xv.x + xv.y*xv.y + xv.z*xv.z + xv.w*xv.w;
#pragma unroll
  for (int m = 16; m >= 1; m >>= 1){ s += __shfl_xor(s, m, 32); q += __shfl_xor(q, m, 32); }
  float mean = s * (1.0f/CCH);
  float var  = q * (1.0f/CCH) - mean*mean;
  float rstd = rsqrtf(var + 1e-5f);
  const float4 sv = *reinterpret_cast<const float4*>(sc + lane*4);
  const float4 bv = *reinterpret_cast<const float4*>(bi + lane*4);
  union { uint2 u; bhalf e[4]; } o;
  o.e[0] = (bhalf)((xv.x-mean)*rstd*sv.x + bv.x);
  o.e[1] = (bhalf)((xv.y-mean)*rstd*sv.y + bv.y);
  o.e[2] = (bhalf)((xv.z-mean)*rstd*sv.z + bv.z);
  o.e[3] = (bhalf)((xv.w-mean)*rstd*sv.w + bv.w);
  *reinterpret_cast<uint2*>(out + (size_t)row*CCH + lane*4) = o.u;
}

// ------- weight f32 (K x Nc, row stride) -> bf16 TRANSPOSED (Nc x K) -------
// All GEMMs then consume B in the K-contiguous fast path (no LDS scatter).
__global__ __launch_bounds__(256) void convT_bf16_kernel(const float* __restrict__ src,
    bhalf* __restrict__ dst, int K, int Nc, int src_stride)
{
  int i = blockIdx.x * 256 + threadIdx.x;
  if (i >= K*Nc) return;
  int n = i / K, k = i - n*K;
  dst[i] = (bhalf)src[(size_t)k*src_stride + n];
}

// ---------------- channel LN over C (axis 0 of (C,N,N)) -> bf16 (M2,C) ----------------
__global__ __launch_bounds__(256) void chnorm_kernel(const float* __restrict__ obuf,
    const float* __restrict__ cs, const float* __restrict__ cb, bhalf* __restrict__ tbuf)
{
  __shared__ float tile[CCH][33];
  __shared__ float red[32][8][2];
  __shared__ float stat[32][2];
  int t  = threadIdx.x;
  int m0 = blockIdx.x * 32;
  for (int c = t >> 5; c < CCH; c += 8)
    tile[c][t & 31] = obuf[(size_t)c*M2 + m0 + (t & 31)];
  __syncthreads();
  int j = t >> 3, part = t & 7;
  float s = 0.0f, q = 0.0f;
#pragma unroll
  for (int c = part*16; c < part*16+16; c++){ float v = tile[c][j]; s += v; q += v*v; }
  red[j][part][0] = s; red[j][part][1] = q;
  __syncthreads();
  if (part == 0){
    float ss=0.0f, qq=0.0f;
#pragma unroll
    for (int p=0;p<8;p++){ ss += red[j][p][0]; qq += red[j][p][1]; }
    float mean = ss * (1.0f/CCH);
    float var  = qq * (1.0f/CCH) - mean*mean;
    stat[j][0] = mean; stat[j][1] = rsqrtf(var + 1e-5f);
  }
  __syncthreads();
  float mean = stat[j][0], rstd = stat[j][1];
  for (int c = part*16; c < part*16+16; c++)
    tbuf[(size_t)(m0+j)*CCH + c] = (bhalf)((tile[c][j]-mean)*rstd*cs[c] + cb[c]);
}

// ---------------- attention bias: hbias[h][q*N+k] = sum_c x[q,k,c]*Wb[c,h] ----------------
__global__ __launch_bounds__(256) void attn_bias_kernel(const bhalf* __restrict__ xln,
    const float* __restrict__ Wb, float* __restrict__ hbias)
{
  __shared__ float w[CCH][4];
  int t = threadIdx.x;
  if (t < CCH){
#pragma unroll
    for (int h=0;h<4;h++) w[t][h] = Wb[t*4 + h];
  }
  __syncthreads();
  size_t m = (size_t)blockIdx.x * 256 + t;
  float acc[4] = {0,0,0,0};
  for (int c = 0; c < CCH; c += 8){
    union { uint4 u; bhalf e[8]; } xv;
    xv.u = *reinterpret_cast<const uint4*>(xln + m*CCH + c);
#pragma unroll
    for (int i=0;i<8;i++){ float x = (float)xv.e[i];
#pragma unroll
      for (int h=0;h<4;h++) acc[h] += x * w[c+i][h];
    }
  }
#pragma unroll
  for (int h=0;h<4;h++) hbias[(size_t)h*M2 + m] = acc[h];
}

// ---------------- generic WMMA GEMM, double-buffered LDS, fused epilogues ----------------
// B is always (Nc x K) row-major (weights pre-transposed at conversion time).
enum { EPI_F32=0, EPI_BF16=1, EPI_TRI=2, EPI_GATE_RES=3, EPI_GATEMUL=4, EPI_RES=5, EPI_SWISHMUL=6 };

__global__ __launch_bounds__(256) void gemm_bf16_kernel(
    const bhalf* __restrict__ A, int a_trans,
    const bhalf* __restrict__ B,
    int M, int Nc, int K,
    long long sA, long long sB, long long sO,
    int epi,
    const float* __restrict__ bias,
    const bhalf* __restrict__ aux,
    const float* __restrict__ mask,
    float* out_f32, bhalf* out_bf16,
    bhalf* outA, bhalf* outB, int tri_mode,
    const float* __restrict__ rbase, float* rdest,
    int store_trans, float scale)
{
  __shared__ __align__(16) bhalf As[2][128][40];
  __shared__ __align__(16) bhalf Bs[2][64][40];
  int z = blockIdx.z;
  A += (size_t)z * sA;
  B += (size_t)z * sB;
  if (out_f32) out_f32 += (size_t)z * sO;
  int m0 = blockIdx.y * 128, n0 = blockIdx.x * 64;
  int t = threadIdx.x, wv = t >> 5, lane = t & 31;
  int wm = (wv & 3) * 32, wn = (wv >> 2) * 32;
  int l16 = lane & 15, lh = lane >> 4;

  // hoisted, loop-invariant source pointers
  int ar = t >> 1, ak = (t & 1) * 16;
  int am = m0 + ar;
  int arow = a_trans ? ((am % NSEQ) * NSEQ + (am / NSEQ)) : am;
  const bhalf* Ap = A + (size_t)arow * K + ak;
  int br = t >> 2, bc = (t & 3) * 8;
  const bhalf* Bp = B + (size_t)(n0 + br) * K + bc;

  // prologue: stage K-tile 0
  uint4 areg = *reinterpret_cast<const uint4*>(Ap);
  uint4 breg = *reinterpret_cast<const uint4*>(Bp);
  *reinterpret_cast<uint4*>(&As[0][ar][ak]) = areg;
  *reinterpret_cast<uint4*>(&Bs[0][br][bc]) = breg;
  __syncthreads();

  v8f acc[2][2];
  acc[0][0]=vz8(); acc[0][1]=vz8(); acc[1][0]=vz8(); acc[1][1]=vz8();
  int nk = K >> 5;
  for (int it = 0; it < nk; it++){
    int cur = it & 1;
    bool more = (it + 1 < nk);
    if (more){                       // prefetch next tile while WMMAs run
      areg = *reinterpret_cast<const uint4*>(Ap + ((size_t)(it+1) << 5));
      breg = *reinterpret_cast<const uint4*>(Bp + ((size_t)(it+1) << 5));
    }
    v16bf af0 = ld_frag_a(&As[cur][wm     ][0], 40);
    v16bf af1 = ld_frag_a(&As[cur][wm + 16][0], 40);
    v16bf bf0 = ld_frag_b(&Bs[cur][wn     ][0], 40);
    v16bf bf1 = ld_frag_b(&Bs[cur][wn + 16][0], 40);
    acc[0][0] = wmma_bf16(af0, bf0, acc[0][0]);
    acc[0][1] = wmma_bf16(af0, bf1, acc[0][1]);
    acc[1][0] = wmma_bf16(af1, bf0, acc[1][0]);
    acc[1][1] = wmma_bf16(af1, bf1, acc[1][1]);
    if (more){
      int nxt = cur ^ 1;
      *reinterpret_cast<uint4*>(&As[nxt][ar][ak]) = areg;
      *reinterpret_cast<uint4*>(&Bs[nxt][br][bc]) = breg;
      __syncthreads();
    }
  }

#pragma unroll
  for (int i=0;i<2;i++)
#pragma unroll
  for (int j=0;j<2;j++)
#pragma unroll
  for (int g=0;g<8;g++){
    int m = m0 + wm + i*16 + lh*8 + g;
    int n = n0 + wn + j*16 + l16;
    float v = acc[i][j][g];
    switch (epi){
      case EPI_F32:
        out_f32[(size_t)m*Nc + n] = v; break;
      case EPI_BF16: {
        float r = v * scale + (bias ? bias[n] : 0.0f);
        out_bf16[(size_t)m*Nc + n] = (bhalf)r;
      } break;
      case EPI_TRI: {
        float gate = sigmf((float)aux[(size_t)m*Nc + n]);
        float val  = gate * v * mask[m];
        int c = n >> 1, sb = n & 1;
        int ii = m / NSEQ, jj = m - ii*NSEQ;
        bhalf bv = (bhalf)val;
        if (tri_mode == 0) (sb ? outB : outA)[((size_t)c*NSEQ + ii)*NSEQ + jj] = bv;
        else               (sb ? outA : outB)[((size_t)c*NSEQ + jj)*NSEQ + ii] = bv;
      } break;
      case EPI_GATE_RES: {
        float y = v + bias[n];
        float gate = sigmf((float)aux[(size_t)m*Nc + n]);
        rdest[(size_t)m*Nc + n] = rbase[(size_t)m*Nc + n] + y * gate;
      } break;
      case EPI_GATEMUL: {
        float gate = sigmf(v + bias[n]);
        out_bf16[(size_t)m*Nc + n] = (bhalf)((float)aux[(size_t)m*Nc + n] * gate);
      } break;
      case EPI_RES: {
        float y = v + bias[n];
        int dr = store_trans ? ((m % NSEQ)*NSEQ + (m / NSEQ)) : m;
        rdest[(size_t)dr*Nc + n] = rbase[(size_t)dr*Nc + n] + y;
      } break;
      case EPI_SWISHMUL: {
        float zz = (float)aux[(size_t)m*Nc + n];
        out_bf16[(size_t)m*Nc + n] = (bhalf)(v * zz * sigmf(zz));
      } break;
    }
  }
}

// ---------------- flash attention: (b,h) batched, 128-q x 64-k tiles ----------------
__global__ __launch_bounds__(256) void flash_attn_kernel(
    const bhalf* __restrict__ qbuf, const bhalf* __restrict__ kbuf, const bhalf* __restrict__ vbuf,
    const float* __restrict__ hbias, const float* __restrict__ pm, bhalf* __restrict__ wa)
{
  __shared__ __align__(16) bhalf Qs[128][40];
  __shared__ __align__(16) bhalf Ks[2][64][40];
  __shared__ __align__(16) bhalf Vt[2][32][72];   // V transposed [d][k]
  __shared__ __align__(16) bhalf Ps[8][16][72];   // per-wave P tile
  int qt = blockIdx.x, h = blockIdx.y, b = blockIdx.z;
  int q0 = qt * 128;
  int t = threadIdx.x, wv = t >> 5, lane = t & 31;
  int l16 = lane & 15, lh = lane >> 4;
  int kr = t >> 2, kc = (t & 3) * 8;
  const bhalf* kp = kbuf + ((size_t)b*NSEQ + kr)*HD + h*32 + kc;
  const bhalf* vp = vbuf + ((size_t)b*NSEQ + kr)*HD + h*32 + kc;
  {
    int r = t >> 1, ko = (t & 1) * 16;
    *reinterpret_cast<uint4*>(&Qs[r][ko]) =
      *reinterpret_cast<const uint4*>(qbuf + ((size_t)b*NSEQ + q0 + r)*HD + h*32 + ko);
  }
  // stage K/V tile 0
  uint4 kreg = *reinterpret_cast<const uint4*>(kp);
  union { uint4 u; bhalf e[8]; } vreg;
  vreg.u = *reinterpret_cast<const uint4*>(vp);
  *reinterpret_cast<uint4*>(&Ks[0][kr][kc]) = kreg;
#pragma unroll
  for (int i=0;i<8;i++) Vt[0][kc+i][kr] = vreg.e[i];
  __syncthreads();

  v16bf qf = ld_frag_a(&Qs[wv*16][0], 40);
  float m_i[8], l_i[8];
#pragma unroll
  for (int g=0; g<8; g++){ m_i[g] = -1e30f; l_i[g] = 0.0f; }
  v8f o0 = vz8(), o1 = vz8();
  const int nkt = NSEQ / 64;
  for (int it = 0; it < nkt; it++){
    int kt  = it * 64;
    int cur = it & 1;
    bool more = (it + 1 < nkt);
    if (more){
      kreg   = *reinterpret_cast<const uint4*>(kp + (size_t)(it+1)*64*HD);
      vreg.u = *reinterpret_cast<const uint4*>(vp + (size_t)(it+1)*64*HD);
    }
    v8f s[4];
#pragma unroll
    for (int nt=0; nt<4; nt++){
      v16bf kf = ld_frag_b(&Ks[cur][nt*16][0], 40);
      s[nt] = wmma_bf16(qf, kf, vz8());
    }
    int qb = q0 + wv*16 + lh*8;
#pragma unroll
    for (int nt=0; nt<4; nt++){
      int kidx = kt + nt*16 + l16;
      float mk = pm[(size_t)kidx*NSEQ + b];
#pragma unroll
      for (int g=0; g<8; g++){
        float val = s[nt][g] + hbias[(size_t)h*M2 + (size_t)(qb+g)*NSEQ + kidx];
        s[nt][g] = (mk > 0.5f) ? val : -1.0e9f;
      }
    }
    float rmax[8], rsum[8], corr[8];
#pragma unroll
    for (int g=0; g<8; g++){
      float mx = s[0][g];
#pragma unroll
      for (int nt=1; nt<4; nt++) mx = fmaxf(mx, s[nt][g]);
#pragma unroll
      for (int msk=8; msk>=1; msk>>=1) mx = fmaxf(mx, __shfl_xor(mx, msk, 32));
      rmax[g] = mx;
    }
#pragma unroll
    for (int g=0; g<8; g++){
      float mn = fmaxf(m_i[g], rmax[g]);
      corr[g] = __expf(m_i[g] - mn);
      m_i[g] = mn;
      rsum[g] = 0.0f;
    }
#pragma unroll
    for (int nt=0; nt<4; nt++)
#pragma unroll
    for (int g=0; g<8; g++){
      float p = __expf(s[nt][g] - m_i[g]);
      s[nt][g] = p;
      rsum[g] += p;
    }
#pragma unroll
    for (int g=0; g<8; g++){
#pragma unroll
      for (int msk=8; msk>=1; msk>>=1) rsum[g] += __shfl_xor(rsum[g], msk, 32);
      l_i[g] = l_i[g]*corr[g] + rsum[g];
      o0[g] *= corr[g]; o1[g] *= corr[g];
    }
#pragma unroll
    for (int nt=0; nt<4; nt++)
#pragma unroll
    for (int g=0; g<8; g++)
      Ps[wv][lh*8 + g][nt*16 + l16] = (bhalf)s[nt][g];
    __syncthreads();   // Ps visible; all waves done with S-phase reads of Ks[cur]
#pragma unroll
    for (int kh=0; kh<2; kh++){
      v16bf pf = ld_frag_a(&Ps[wv][0][kh*32], 72);
      v16bf v0 = ld_frag_b(&Vt[cur][0][kh*32], 72);
      v16bf v1 = ld_frag_b(&Vt[cur][16][kh*32], 72);
      o0 = wmma_bf16(pf, v0, o0);
      o1 = wmma_bf16(pf, v1, o1);
    }
    if (more){
      int nxt = cur ^ 1;
      *reinterpret_cast<uint4*>(&Ks[nxt][kr][kc]) = kreg;
#pragma unroll
      for (int i=0;i<8;i++) Vt[nxt][kc+i][kr] = vreg.e[i];
      __syncthreads();
    }
  }
#pragma unroll
  for (int g=0; g<8; g++){
    float inv = 1.0f / l_i[g];
    size_t row = (size_t)b*NSEQ + q0 + wv*16 + lh*8 + g;
    wa[row*HD + h*32 + l16]      = (bhalf)(o0[g]*inv);
    wa[row*HD + h*32 + 16 + l16] = (bhalf)(o1[g]*inv);
  }
}

// ------------------------------ host orchestration ------------------------------
static inline void launch_gemm(hipStream_t st, const bhalf* A, int aT, const bhalf* B,
    int M, int Nc, int K, int batch, long long sA, long long sB, long long sO,
    int epi, const float* bias, const bhalf* aux, const float* mask,
    float* of32, bhalf* obf, bhalf* oA, bhalf* oB, int tri,
    const float* rbase, float* rdest, int strans, float scale)
{
  dim3 g(Nc/64, M/128, batch);
  gemm_bf16_kernel<<<g, 256, 0, st>>>(A,aT,B,M,Nc,K,sA,sB,sO,epi,bias,aux,mask,
                                      of32,obf,oA,oB,tri,rbase,rdest,strans,scale);
}

extern "C" void kernel_launch(void* const* d_in, const int* in_sizes, int n_in,
                              void* d_out, int out_size, void* d_ws, size_t ws_size,
                              hipStream_t stream)
{
  (void)in_sizes; (void)n_in; (void)out_size; (void)ws_size;
  const float* act_in = (const float*)d_in[0];
  const float* pmask  = (const float*)d_in[1];
  float* act = (float*)d_out;            // working activation lives in d_out

  // workspace layout (~342 MB): [xln bf16 | weight slab | stage slab (302MB)]
  char* ws  = (char*)d_ws;
  bhalf* xln = (bhalf*)ws;
  bhalf* wsl = (bhalf*)(ws + (size_t)M2*CCH*2);
  char*  S   = ws + (size_t)M2*CCH*2 + (size_t)2*1024*1024;

  bhalf* w_tm[2][4]; size_t wo = 0;
  for (int s_=0; s_<2; s_++){
    w_tm[s_][0] = wsl + wo; wo += 32768;  // Wp^T
    w_tm[s_][1] = wsl + wo; wo += 32768;  // Wg^T
    w_tm[s_][2] = wsl + wo; wo += 16384;  // Wo^T
    w_tm[s_][3] = wsl + wo; wo += 16384;  // Wgl^T
  }
  bhalf* w_at[2][5];
  for (int s_=0; s_<2; s_++)
    for (int i=0;i<5;i++){ w_at[s_][i] = wsl + wo; wo += 16384; } // Wq,Wk,Wv,Wgate,Wo (^T)
  bhalf* w1a = wsl + wo; wo += 65536;
  bhalf* w1b = wsl + wo; wo += 65536;
  bhalf* w2  = wsl + wo; wo += 65536;

  auto cvtT = [&](const void* src, bhalf* dst, int K, int Nc, int stride){
    int tot = K*Nc;
    convT_bf16_kernel<<<dim3((tot+255)/256), 256, 0, stream>>>((const float*)src, dst, K, Nc, stride);
  };
  const int tmbase[2] = {2, 12};
  const int atbase[2] = {22, 33};
  for (int s_=0;s_<2;s_++){
    int p = tmbase[s_];
    cvtT(d_in[p+2], w_tm[s_][0], 128, 256, 256);
    cvtT(d_in[p+3], w_tm[s_][1], 128, 256, 256);
    cvtT(d_in[p+6], w_tm[s_][2], 128, 128, 128);
    cvtT(d_in[p+8], w_tm[s_][3], 128, 128, 128);
  }
  for (int s_=0;s_<2;s_++){
    int p = atbase[s_];
    cvtT(d_in[p+2], w_at[s_][0], 128, 128, 128);
    cvtT(d_in[p+3], w_at[s_][1], 128, 128, 128);
    cvtT(d_in[p+4], w_at[s_][2], 128, 128, 128);
    cvtT(d_in[p+7], w_at[s_][3], 128, 128, 128);
    cvtT(d_in[p+9], w_at[s_][4], 128, 128, 128);
  }
  cvtT(d_in[46], w1a, 128, 512, 1024);
  cvtT((const float*)d_in[46] + 512, w1b, 128, 512, 1024);
  cvtT(d_in[47], w2, 512, 128, 128);

  // -------- triangular multiplication stages (outgoing, incoming) --------
  for (int s_=0; s_<2; s_++){
    int p = tmbase[s_];
    const float* baseAct = (s_ == 0) ? act_in : act;
    bhalf* gbuf  = (bhalf*)S;
    bhalf* glbuf = (bhalf*)(S + (size_t)M2*256*2);
    bhalf* Abuf  = (bhalf*)(S + (size_t)M2*256*2 + (size_t)M2*128*2);
    bhalf* Bbuf  = Abuf + (size_t)M2*128;
    float* obuf  = (float*)((char*)(Bbuf + (size_t)M2*128));
    bhalf* tbuf  = (bhalf*)((char*)obuf + (size_t)M2*128*4);

    ln_kernel<<<dim3(M2/8), 256, 0, stream>>>(baseAct, (const float*)d_in[p], (const float*)d_in[p+1], xln);
    launch_gemm(stream, xln,0, w_tm[s_][1], M2,256,128, 1,0,0,0,
                EPI_BF16, nullptr, nullptr, nullptr, nullptr, gbuf, nullptr,nullptr,0, nullptr,nullptr,0, 1.0f);
    launch_gemm(stream, xln,0, w_tm[s_][0], M2,256,128, 1,0,0,0,
                EPI_TRI, nullptr, gbuf, pmask, nullptr,nullptr, Abuf,Bbuf, s_, nullptr,nullptr,0, 1.0f);
    launch_gemm(stream, Abuf,0, Bbuf, NSEQ,NSEQ,NSEQ, 128,
                (long long)NSEQ*NSEQ,(long long)NSEQ*NSEQ,(long long)NSEQ*NSEQ,
                EPI_F32, nullptr,nullptr,nullptr, obuf, nullptr,nullptr,nullptr,0, nullptr,nullptr,0, 1.0f);
    chnorm_kernel<<<dim3(M2/32), 256, 0, stream>>>(obuf, (const float*)d_in[p+4], (const float*)d_in[p+5], tbuf);
    launch_gemm(stream, xln,0, w_tm[s_][3], M2,128,128, 1,0,0,0,
                EPI_BF16, (const float*)d_in[p+9], nullptr,nullptr, nullptr, glbuf, nullptr,nullptr,0,
                nullptr,nullptr,0, 1.0f);
    launch_gemm(stream, tbuf,0, w_tm[s_][2], M2,128,128, 1,0,0,0,
                EPI_GATE_RES, (const float*)d_in[p+7], glbuf, nullptr,
                nullptr,nullptr,nullptr,nullptr,0, baseAct, act, 0, 1.0f);
  }

  // -------- grid attention stages (row, column) --------
  for (int s_=0; s_<2; s_++){
    int p = atbase[s_];
    int T = s_;
    bhalf* qb  = (bhalf*)S;
    bhalf* kb  = qb  + (size_t)M2*128;
    bhalf* vb  = kb  + (size_t)M2*128;
    bhalf* wab = vb  + (size_t)M2*128;
    bhalf* gwa = wab + (size_t)M2*128;
    float* hbias = (float*)(gwa + (size_t)M2*128);

    ln_kernel<<<dim3(M2/8),256,0,stream>>>(act, (const float*)d_in[p], (const float*)d_in[p+1], xln);
    attn_bias_kernel<<<dim3(M2/256),256,0,stream>>>(xln, (const float*)d_in[p+6], hbias);
    launch_gemm(stream, xln,T, w_at[s_][0], M2,128,128, 1,0,0,0,
                EPI_BF16, nullptr, nullptr,nullptr, nullptr, qb, nullptr,nullptr,0,
                nullptr,nullptr,0, 0.17677669529663689f); // D^-0.5
    launch_gemm(stream, xln,T, w_at[s_][1], M2,128,128, 1,0,0,0,
                EPI_BF16, nullptr, nullptr,nullptr, nullptr, kb, nullptr,nullptr,0, nullptr,nullptr,0, 1.0f);
    launch_gemm(stream, xln,T, w_at[s_][2], M2,128,128, 1,0,0,0,
                EPI_BF16, (const float*)d_in[p+5], nullptr,nullptr, nullptr, vb, nullptr,nullptr,0,
                nullptr,nullptr,0, 1.0f);
    flash_attn_kernel<<<dim3(3,4,NSEQ),256,0,stream>>>(qb, kb, vb, hbias, pmask, wab);
    launch_gemm(stream, xln,T, w_at[s_][3], M2,128,128, 1,0,0,0,
                EPI_GATEMUL, (const float*)d_in[p+8], wab, nullptr, nullptr, gwa, nullptr,nullptr,0,
                nullptr,nullptr,0, 1.0f);
    launch_gemm(stream, gwa,0, w_at[s_][4], M2,128,128, 1,0,0,0,
                EPI_RES, (const float*)d_in[p+10], nullptr,nullptr,
                nullptr,nullptr,nullptr,nullptr,0, act, act, T, 1.0f);
  }

  // -------- transition (SwiGLU-style) --------
  {
    bhalf* g1 = (bhalf*)S;
    bhalf* hb = g1 + (size_t)M2*512;
    ln_kernel<<<dim3(M2/8),256,0,stream>>>(act, (const float*)d_in[44], (const float*)d_in[45], xln);
    launch_gemm(stream, xln,0, w1a, M2,512,128, 1,0,0,0,
                EPI_BF16, nullptr,nullptr,nullptr, nullptr, g1, nullptr,nullptr,0, nullptr,nullptr,0, 1.0f);
    launch_gemm(stream, xln,0, w1b, M2,512,128, 1,0,0,0,
                EPI_SWISHMUL, nullptr, g1, nullptr, nullptr, hb, nullptr,nullptr,0, nullptr,nullptr,0, 1.0f);
    launch_gemm(stream, hb,0, w2, M2,128,512, 1,0,0,0,
                EPI_RES, (const float*)d_in[48], nullptr,nullptr,
                nullptr,nullptr,nullptr,nullptr,0, act, act, 0, 1.0f);
  }
}
